// GNNNodeClassifier_16363825398631
// MI455X (gfx1250) — matrix operand order repaired and checked by reading, not compiled, not run on hardware
//
#include <hip/hip_runtime.h>
#include <hip/hip_bf16.h>
#include <math.h>

typedef __attribute__((ext_vector_type(16))) _Float16 v16h;
typedef __attribute__((ext_vector_type(8)))  float    v8f;

#define NNODES 100000
#define NEDGES 1600000
#define D_IN   128
#define D_HID  128
#define D_OUT  64
#define KDIM   128
#define LDS_STRIDE 136   // 128 + 8 halves; 272B row stride -> conflict-free frag reads

// -------------------------------------------------- CDNA5 async-copy helpers
// global_load_async_to_lds_b128: VDST = VGPR with LDS byte address, VADDR =
// 64-bit global address (GV mode). Tracked by ASYNCcnt. LDS offset is the low
// 32 bits of the generic pointer.
__device__ __forceinline__ void async_ld_b128(unsigned lds_byte, const void* gaddr) {
    asm volatile("global_load_async_to_lds_b128 %0, %1, off"
                 :: "v"(lds_byte), "v"(gaddr) : "memory");
}

__device__ __forceinline__ void wait_async0() {
#if __has_builtin(__builtin_amdgcn_s_wait_asynccnt)
    __builtin_amdgcn_s_wait_asynccnt(0);
#else
    asm volatile("s_wait_asynccnt 0x0" ::: "memory");
#endif
}

// ---------------------------------------------------------------- utilities
__global__ void k_fill(float* p, int n, float v) {
    int i = blockIdx.x * blockDim.x + threadIdx.x;
    if (i < n) p[i] = v;
}

__global__ void k_zero(float* p, int n) {
    int i = blockIdx.x * blockDim.x + threadIdx.x;
    if (i < n) p[i] = 0.0f;
}

// W[K x N] f32 -> Wt[N x K] f16 (transposed, so B fragments are contiguous)
__global__ void k_w_to_f16_t(const float* __restrict__ W, _Float16* __restrict__ Wt,
                             int K, int N) {
    int idx = blockIdx.x * blockDim.x + threadIdx.x;
    if (idx < K * N) {
        int k = idx / N;
        int n = idx - k * N;
        Wt[n * K + k] = (_Float16)W[idx];
    }
}

// deg[col[e]] += w[e]   (deg pre-filled with 1.0 for the self loop)
__global__ void k_deg_accum(const int* __restrict__ col, const float* __restrict__ w,
                            float* __restrict__ deg, int nedges) {
    int e = blockIdx.x * blockDim.x + threadIdx.x;
    if (e < nedges) atomicAdd(&deg[col[e]], w[e]);
}

__global__ void k_rsqrt_inplace(float* __restrict__ d, int n) {
    int i = blockIdx.x * blockDim.x + threadIdx.x;
    if (i < n) {
        float v = d[i];
        d[i] = (v > 0.0f) ? rsqrtf(v) : 0.0f;
    }
}

// ---------------------------------------------------------------- WMMA GEMM
// C[M x NCOLS] = act(A[M x 128] @ W[128 x NCOLS] (+ bias))
// A is f32 (layer 1: x) or f16 (layers 2/3: activations). Wt is f16 transposed
// weight [NCOLS x 128], async-staged into padded LDS once per block.
// 256 threads = 8 waves; K = 128 -> 4 v_wmma_f32_16x16x32_f16 per wave.
union frag_cast { uint4 q[2]; v16h h; };

template<typename AT, int NCOLS, bool SIGMOID>
__global__ void k_gemm_wmma(const AT* __restrict__ A, const _Float16* __restrict__ Wt,
                            const float* __restrict__ bias, void* __restrict__ Cout) {
    __shared__ _Float16 sW[NCOLS * LDS_STRIDE];

    // --- async-stage Wt into LDS (16B chunks), then wait + barrier ---
    {
        const uint4* g = (const uint4*)Wt;               // [NCOLS][16] uint4 rows
        for (int i = threadIdx.x; i < NCOLS * 16; i += 256) {
            int row = i >> 4;
            int c8  = i & 15;
            unsigned lds_byte =
                (unsigned)(uintptr_t)(void*)&sW[row * LDS_STRIDE + (c8 << 3)];
            async_ld_b128(lds_byte, g + i);
        }
        wait_async0();
    }
    __syncthreads();

    const int lane      = threadIdx.x & 31;
    const int wave      = threadIdx.x >> 5;
    const int col_tiles = NCOLS / 16;
    const int ct        = wave % col_tiles;
    const int rt        = wave / col_tiles;
    const int rows_blk  = 16 * (8 / col_tiles);
    const int row0      = blockIdx.x * rows_blk + rt * 16;  // M divisible by rows_blk
    const int half_id   = lane >> 4;
    const int mn        = lane & 15;

    const AT* arow = A + (size_t)(row0 + mn) * KDIM;
    const _Float16* brow = &sW[(ct * 16 + mn) * LDS_STRIDE + half_id * 16];

    v8f acc = {};
    #pragma unroll
    for (int kt = 0; kt < KDIM; kt += 32) {
        // A frag (16-bit, 16x32): lane -> row, e -> K = kt + (e>>3)*16 + half*8 + (e&7)
        frag_cast a;
        if constexpr (sizeof(AT) == 2) {
            // f16 A: two contiguous 16B runs per lane per k-tile
            const uint4* ap = (const uint4*)arow;
            a.q[0] = ap[(kt >> 3) + half_id];        // K = kt+half*8 .. +7
            a.q[1] = ap[(kt >> 3) + 2 + half_id];    // K = kt+16+half*8 .. +7
        } else {
            #pragma unroll
            for (int e = 0; e < 16; ++e) {
                int ka = kt + ((e >> 3) << 4) + (half_id << 3) + (e & 7);
                a.h[e] = (_Float16)arow[ka];
            }
        }
        // B frag (32x16): lane%16 -> col, e -> K = kt + (e>>3)*16 + half*8 + (e&7)
        frag_cast b;
        b.q[0] = *(const uint4*)&brow[kt - half_id * 8];
        b.q[1] = *(const uint4*)&brow[kt + 16 - half_id * 8];
        acc = __builtin_amdgcn_wmma_f32_16x16x32_f16(
                  false, a.h, false, b.h, (short)0, acc, false, false);
    }

    #pragma unroll
    for (int r = 0; r < 8; ++r) {
        int m = row0 + r + (half_id << 3);
        int n = (ct << 4) + mn;
        if constexpr (SIGMOID) {
            float v = acc[r] + bias[n];
            ((float*)Cout)[(size_t)m * NCOLS + n] = 1.0f / (1.0f + __expf(-v));
        } else {
            ((_Float16*)Cout)[(size_t)m * NCOLS + n] = (_Float16)acc[r];
        }
    }
}

// ---------------------------------------------------------------- scatter
// agg[col[e]] += dinv[row]*w*dinv[col] * t[row]   (one edge per wave; t is f16
// so the gather is 256B/edge; 4 consecutive dims per lane, f32 atomics)
__global__ void k_aggregate(const int* __restrict__ row, const int* __restrict__ col,
                            const float* __restrict__ w, const float* __restrict__ dinv,
                            const _Float16* __restrict__ t, float* __restrict__ agg,
                            int nedges) {
    int e = blockIdx.x * (blockDim.x >> 5) + (threadIdx.x >> 5);
    if (e >= nedges) return;
    int lane = threadIdx.x & 31;
    int r = row[e];
    int c = col[e];
    float norm = dinv[r] * w[e] * dinv[c];
    const _Float16* src = t   + (size_t)r * D_HID;
    float*          dst = agg + (size_t)c * D_HID;
    union { uint2 u; _Float16 h[4]; } sv;
    sv.u = *(const uint2*)&src[lane << 2];        // 8B-aligned vector load
    #pragma unroll
    for (int j = 0; j < 4; ++j) {
        atomicAdd(&dst[(lane << 2) + j], norm * (float)sv.h[j]);
    }
}

// h = (f16) relu(agg + dinv^2 * t + b)   (self-loop folded in)
__global__ void k_finalize(const float* __restrict__ agg, const _Float16* __restrict__ t,
                           const float* __restrict__ dinv, const float* __restrict__ bias,
                           _Float16* __restrict__ h, int total) {
    int idx = blockIdx.x * blockDim.x + threadIdx.x;
    if (idx >= total) return;
    int nidx = idx >> 7;       // /128
    int d    = idx & 127;
    float di = dinv[nidx];
    float v  = agg[idx] + di * di * (float)t[idx] + bias[d];
    h[idx] = (_Float16)(v > 0.0f ? v : 0.0f);
}

// ---------------------------------------------------------------- launcher
extern "C" void kernel_launch(void* const* d_in, const int* in_sizes, int n_in,
                              void* d_out, int out_size, void* d_ws, size_t ws_size,
                              hipStream_t stream) {
    const float* x   = (const float*)d_in[0];
    const int*   ei  = (const int*)  d_in[1];   // [2, E]: row = ei[e], col = ei[E+e]
    const float* ew  = (const float*)d_in[2];
    const float* W1  = (const float*)d_in[3];
    const float* b1  = (const float*)d_in[4];
    const float* W2  = (const float*)d_in[5];
    const float* b2  = (const float*)d_in[6];
    const float* Wfc = (const float*)d_in[7];
    const float* bfc = (const float*)d_in[8];
    float* out = (float*)d_out;

    const int* row = ei;
    const int* col = ei + NEDGES;

    // workspace layout (bytes, 256-aligned)
    char* ws = (char*)d_ws;
    const size_t f16Bytes = (size_t)NNODES * D_HID * sizeof(_Float16); // 25,600,000
    const size_t f32Bytes = (size_t)NNODES * D_HID * sizeof(float);    // 51,200,000
    _Float16* bufT  = (_Float16*)(ws);                           // t = h @ W (f16)
    float*    bufA  = (float*)(ws + f16Bytes);                   // agg (f32 atomics)
    _Float16* bufH  = (_Float16*)(ws + f16Bytes + f32Bytes);     // h (f16)
    float*    dinv  = (float*)(ws + 2 * f16Bytes + f32Bytes);
    char*     wbase = ws + 2 * f16Bytes + f32Bytes
                         + (((size_t)NNODES * 4 + 255) & ~(size_t)255);
    _Float16* W1t   = (_Float16*)(wbase);                         // [128][128]
    _Float16* W2t   = (_Float16*)(wbase + 128 * 128 * 2);         // [128][128]
    _Float16* Wfct  = (_Float16*)(wbase + 2 * 128 * 128 * 2);     // [64][128]

    const int feat_total = NNODES * D_HID;

    // 1) degree (self loop = 1.0) -> dinv
    k_fill<<<(NNODES + 255) / 256, 256, 0, stream>>>(dinv, NNODES, 1.0f);
    k_deg_accum<<<(NEDGES + 255) / 256, 256, 0, stream>>>(col, ew, dinv, NEDGES);
    k_rsqrt_inplace<<<(NNODES + 255) / 256, 256, 0, stream>>>(dinv, NNODES);

    // 2) weights -> transposed f16
    k_w_to_f16_t<<<(128 * 128 + 255) / 256, 256, 0, stream>>>(W1, W1t, 128, 128);
    k_w_to_f16_t<<<(128 * 128 + 255) / 256, 256, 0, stream>>>(W2, W2t, 128, 128);
    k_w_to_f16_t<<<(128 * 64 + 255) / 256, 256, 0, stream>>>(Wfc, Wfct, 128, 64);

    // 3) layer 1: t = x @ W1 ; agg ; finalize -> h1 (f16) in bufH
    k_gemm_wmma<float, 128, false><<<NNODES / 16, 256, 0, stream>>>(x, W1t, nullptr, bufT);
    k_zero<<<(feat_total + 255) / 256, 256, 0, stream>>>(bufA, feat_total);
    k_aggregate<<<NEDGES / 8, 256, 0, stream>>>(row, col, ew, dinv, bufT, bufA, NEDGES);
    k_finalize<<<(feat_total + 255) / 256, 256, 0, stream>>>(bufA, bufT, dinv, b1, bufH, feat_total);

    // 4) layer 2: t2 = h1 @ W2 (f16 A path) ; agg ; finalize -> h2 in bufH
    k_gemm_wmma<_Float16, 128, false><<<NNODES / 16, 256, 0, stream>>>(bufH, W2t, nullptr, bufT);
    k_zero<<<(feat_total + 255) / 256, 256, 0, stream>>>(bufA, feat_total);
    k_aggregate<<<NEDGES / 8, 256, 0, stream>>>(row, col, ew, dinv, bufT, bufA, NEDGES);
    k_finalize<<<(feat_total + 255) / 256, 256, 0, stream>>>(bufA, bufT, dinv, b2, bufH, feat_total);

    // 5) head: out = sigmoid(h2 @ Wfc + bfc)  (32 rows/block, f32 output)
    k_gemm_wmma<_Float16, 64, true><<<NNODES / 32, 256, 0, stream>>>(bufH, Wfct, bfc, out);
}
